// Attention_13022340841801
// MI455X (gfx1250) — compile-verified, matrix operand order checked by reading, not run on hardware
//
#include <hip/hip_runtime.h>
#include <hip/hip_bf16.h>

// ---------------------------------------------------------------------------
// Attention (B=1, S=2048, DIM=2048, 32 Q heads / 8 KV heads, head_dim=64)
// bf16 WMMA pipeline for gfx1250 (CDNA5), wave32.
//   - QKV / out-proj: block-tiled GEMM, B tile double-buffered in LDS via
//     GLOBAL_LOAD_ASYNC_TO_LDS_B128 (fallback: ds_store), 8 WMMA / wave / k-step
//   - flash attention per (head, 16-row q tile), online softmax, P re-fragment
//     through LDS, causal mask computed inline (mask input never read)
// ---------------------------------------------------------------------------

#define S_LEN 2048
#define DIM_  2048
#define NH    32
#define NKV   8
#define HD    64

typedef __bf16 bf16;
typedef __attribute__((ext_vector_type(16))) __bf16 v16bf;
typedef __attribute__((ext_vector_type(8)))  __bf16 v8bf;
typedef __attribute__((ext_vector_type(8)))  float  v8f;
typedef __attribute__((ext_vector_type(4)))  int    v4i;

// address-space-qualified views for the async-copy builtin
typedef __attribute__((address_space(1))) v4i ga_v4i;   // global
typedef __attribute__((address_space(3))) v4i lds_v4i;  // LDS

#if __has_builtin(__builtin_amdgcn_global_load_async_to_lds_b128)
#define ASYNC_LDS 1
#endif

// 16-byte global -> LDS stage (async when available)
static __device__ inline void stage16(const bf16* g, bf16* l) {
#ifdef ASYNC_LDS
  __builtin_amdgcn_global_load_async_to_lds_b128(
      (ga_v4i*)g, (lds_v4i*)l, 0, 0);
#else
  *(v8bf*)l = *(const v8bf*)g;
#endif
}

static __device__ inline void wait_async_lds() {
#ifdef ASYNC_LDS
#if __has_builtin(__builtin_amdgcn_s_wait_asynccnt)
  __builtin_amdgcn_s_wait_asynccnt(0);
#else
  asm volatile("s_wait_asynccnt 0x0" ::: "memory");
#endif
#endif
}

static __device__ inline v8f wmma_bf16f32(v16bf a, v16bf b, v8f c) {
  // D = A(16x32 bf16) * B(32x16 bf16) + C(16x16 f32)
  return __builtin_amdgcn_wmma_f32_16x16x32_bf16(
      false, a, false, b, (short)0, c, false, false);
}

// Load a 16x32 operand fragment from a row-major bf16 matrix (leading dim ld).
// Per ISA 16-bit A layout: lanes 0-15 hold M=lane, K=0..7 (v0-3) and 16..23
// (v4-7); lanes 16-31 hold K=8..15 and 24..31.  Two 16B contiguous loads/lane.
static __device__ inline v16bf load_frag_g(const bf16* __restrict__ p, int ld,
                                           int row0, int k0, int lane) {
  const int m  = lane & 15;
  const int kh = (lane >> 4) << 3;
  const bf16* base = p + (size_t)(row0 + m) * ld + (k0 + kh);
  v8bf lo = *(const v8bf*)(base);
  v8bf hi = *(const v8bf*)(base + 16);
  v16bf f;
#pragma unroll
  for (int i = 0; i < 8; ++i) { f[i] = lo[i]; f[i + 8] = hi[i]; }
  return f;
}

// Same fragment gather from an LDS-staged tile with ld = 32.
static __device__ inline v16bf load_frag_l(const bf16* p, int row0, int lane) {
  const int m  = lane & 15;
  const int kh = (lane >> 4) << 3;
  const bf16* base = p + (row0 + m) * 32 + kh;
  v16bf f;
#pragma unroll
  for (int i = 0; i < 8; ++i) { f[i] = base[i]; f[i + 8] = base[16 + i]; }
  return f;
}

// ---------------------------------------------------------------------------
// Elementwise pack / convert kernels
// ---------------------------------------------------------------------------
__global__ void k_cvt_bf16(const float* __restrict__ in, bf16* __restrict__ out,
                           int n) {
  int i = blockIdx.x * 256 + threadIdx.x;
  if (i < n) out[i] = (bf16)in[i];
}

// W[K][N] f32  ->  WT[N][K] bf16  (so the GEMM B operand loads contiguously)
__global__ void k_tr_cvt(const float* __restrict__ W, bf16* __restrict__ WT,
                         int K, int N) {
  int t = blockIdx.x * 256 + threadIdx.x;
  if (t >= K * N) return;
  int k = t % K;
  int n = t / K;
  WT[(size_t)n * K + k] = (bf16)W[(size_t)k * N + n];
}

// RoPE + repack f32 [s][nheads*64] -> bf16 [h][s][64]
__global__ void k_rope_pack(const float* __restrict__ X, bf16* __restrict__ out,
                            const float* __restrict__ fc,
                            const float* __restrict__ fs, int nheads) {
  int t = blockIdx.x * 256 + threadIdx.x;       // (h, s, j)
  int total = nheads * S_LEN * (HD / 2);
  if (t >= total) return;
  int j = t % (HD / 2);
  int s = (t / (HD / 2)) % S_LEN;
  int h = t / ((HD / 2) * S_LEN);
  int rowstride = nheads * HD;
  float re = X[(size_t)s * rowstride + h * HD + 2 * j];
  float im = X[(size_t)s * rowstride + h * HD + 2 * j + 1];
  float c  = fc[s * (HD / 2) + j];
  float sn = fs[s * (HD / 2) + j];
  size_t o = ((size_t)h * S_LEN + s) * HD + 2 * j;
  out[o]     = (bf16)(re * c - im * sn);
  out[o + 1] = (bf16)(re * sn + im * c);
}

// V f32 [s][NKV*64] -> bf16 V^T [kv][d][s]  (PV WMMA B-operand layout)
__global__ void k_pack_vT(const float* __restrict__ Vf, bf16* __restrict__ Vt) {
  int t = blockIdx.x * 256 + threadIdx.x;       // (kv, d, s), coalesced in s
  if (t >= NKV * HD * S_LEN) return;
  int s  = t % S_LEN;
  int d  = (t / S_LEN) % HD;
  int kv = t / (S_LEN * HD);
  Vt[t] = (bf16)Vf[(size_t)s * (NKV * HD) + kv * HD + d];
}

// ---------------------------------------------------------------------------
// bf16 GEMM: C[M][N] f32 = A[M][K] * BT[N][K]^T
// Block tile 256x64 (8 waves x 32 rows). The 64x32 B tile is shared by all
// waves: double-buffered in LDS via async global->LDS copies. Each wave does
// 8 WMMAs per k-step (2 A-frags x 4 B-frags).
// ---------------------------------------------------------------------------
__global__ __launch_bounds__(256) void k_gemm(const bf16* __restrict__ A,
                                              const bf16* __restrict__ BT,
                                              float* __restrict__ C, int M,
                                              int N, int K) {
  __shared__ bf16 Bs[2][64 * 32];               // 2 x 4 KB
  const int tid  = threadIdx.x;
  const int lane = tid & 31;
  const int wave = tid >> 5;
  const int m0 = (blockIdx.x * 8 + wave) * 32;
  const int n0 = blockIdx.y * 64;

  // cooperative staging coords: 4 threads x 16B per 64B row of BT
  const int srow = tid >> 2;                    // 0..63
  const int sseg = (tid & 3) * 8;               // bf16 offset within row

  stage16(&BT[(size_t)(n0 + srow) * K + sseg], &Bs[0][srow * 32 + sseg]);

  v8f acc[2][4] = {};
  int buf = 0;
  for (int k0 = 0; k0 < K; k0 += 32, buf ^= 1) {
    wait_async_lds();
    __syncthreads();                            // Bs[buf] ready for all waves
    if (k0 + 32 < K) {
      stage16(&BT[(size_t)(n0 + srow) * K + (k0 + 32) + sseg],
              &Bs[buf ^ 1][srow * 32 + sseg]);
      __builtin_prefetch(A + (size_t)(m0 + lane) * K + (k0 + 32), 0, 1);
    }
    v16bf a0 = load_frag_g(A, K, m0, k0, lane);
    v16bf a1 = load_frag_g(A, K, m0 + 16, k0, lane);
#pragma unroll
    for (int t = 0; t < 4; ++t) {
      v16bf b = load_frag_l(&Bs[buf][0], 16 * t, lane);
      acc[0][t] = wmma_bf16f32(a0, b, acc[0][t]);
      acc[1][t] = wmma_bf16f32(a1, b, acc[1][t]);
    }
    __syncthreads();                            // reads done before buf reuse
  }
  const int cn = lane & 15;
  const int rowB = (lane >> 4) << 3;  // C layout: lanes 0-15 rows 0-7, 16-31 rows 8-15
#pragma unroll
  for (int u = 0; u < 2; ++u)
#pragma unroll
    for (int t = 0; t < 4; ++t)
#pragma unroll
      for (int i = 0; i < 8; ++i)
        C[(size_t)(m0 + 16 * u + rowB + i) * N + (n0 + 16 * t + cn)] =
            acc[u][t][i];
}

// ---------------------------------------------------------------------------
// Flash attention: per wave, one (head, 16-row q-tile). K-blocks of 32.
// Scores: 4 WMMA; PV: 4 WMMA; online softmax via 16-lane shfl_xor reductions.
// ---------------------------------------------------------------------------
__global__ __launch_bounds__(256) void k_flash(const bf16* __restrict__ Qb,
                                               const bf16* __restrict__ Kb,
                                               const bf16* __restrict__ Vt,
                                               bf16* __restrict__ Ob) {
  __shared__ bf16 lds_p[8][16 * 32];
  const int lane = threadIdx.x & 31;
  const int wave = threadIdx.x >> 5;
  const int tile = blockIdx.x * 8 + wave;   // 32 heads * 128 q-tiles = 4096
  const int h  = tile >> 7;
  const int q0 = (tile & 127) << 4;
  const int kv = h >> 2;                    // rep = NH / NKV = 4
  const bf16* Q  = Qb + (size_t)h * S_LEN * HD;
  const bf16* Kh = Kb + (size_t)kv * S_LEN * HD;
  const bf16* Vh = Vt + (size_t)kv * HD * S_LEN;

  const v16bf qf0 = load_frag_g(Q, HD, q0, 0, lane);
  const v16bf qf1 = load_frag_g(Q, HD, q0, 32, lane);

  v8f o[4] = {};
  float mrow[8], lrow[8];
#pragma unroll
  for (int i = 0; i < 8; ++i) { mrow[i] = -1e30f; lrow[i] = 0.f; }

  const int cn   = lane & 15;
  const int rowB = (lane >> 4) << 3;
  const float scale = 0.125f;               // 1/sqrt(64)
  bf16* pl = lds_p[wave];

  for (int k0 = 0; k0 < q0 + 16; k0 += 32) {
    if (k0 + 32 < q0 + 16) {                // hide next K/V block latency
      __builtin_prefetch(Kh + (size_t)(k0 + 32 + (lane & 31)) * HD, 0, 1);
      __builtin_prefetch(Vh + (size_t)(lane & 31) * S_LEN + k0 + 32, 0, 1);
    }
    // --- scores S = Q K^T for a 16x32 block (two 16x16 C tiles) ---
    v8f sc[2] = {};
#pragma unroll
    for (int t = 0; t < 2; ++t) {
      v16bf b0 = load_frag_g(Kh, HD, k0 + 16 * t, 0, lane);
      v16bf b1 = load_frag_g(Kh, HD, k0 + 16 * t, 32, lane);
      sc[t] = wmma_bf16f32(qf0, b0, sc[t]);
      sc[t] = wmma_bf16f32(qf1, b1, sc[t]);
    }
    // --- scale + causal mask (inline; mask input never read) ---
#pragma unroll
    for (int t = 0; t < 2; ++t)
#pragma unroll
      for (int i = 0; i < 8; ++i) {
        int r = q0 + rowB + i;
        int c = k0 + 16 * t + cn;
        float v = sc[t][i] * scale;
        sc[t][i] = (c <= r) ? v : -1e30f;
      }
    // --- row max across the 16 lanes holding each row ---
    float bmax[8];
#pragma unroll
    for (int i = 0; i < 8; ++i) bmax[i] = fmaxf(sc[0][i], sc[1][i]);
#pragma unroll
    for (int d = 0; d < 4; ++d) {
      int off = 1 << d;
#pragma unroll
      for (int i = 0; i < 8; ++i)
        bmax[i] = fmaxf(bmax[i], __shfl_xor(bmax[i], off, 32));
    }
    float alpha[8];
#pragma unroll
    for (int i = 0; i < 8; ++i) {
      float mn = fmaxf(mrow[i], bmax[i]);
      alpha[i] = __expf(mrow[i] - mn);
      mrow[i]  = mn;
    }
    // --- P = exp(S - m), row sums ---
    float bsum[8];
#pragma unroll
    for (int i = 0; i < 8; ++i) bsum[i] = 0.f;
#pragma unroll
    for (int t = 0; t < 2; ++t)
#pragma unroll
      for (int i = 0; i < 8; ++i) {
        float p = __expf(sc[t][i] - mrow[i]);
        sc[t][i] = p;
        bsum[i] += p;
      }
#pragma unroll
    for (int d = 0; d < 4; ++d) {
      int off = 1 << d;
#pragma unroll
      for (int i = 0; i < 8; ++i)
        bsum[i] += __shfl_xor(bsum[i], off, 32);
    }
#pragma unroll
    for (int i = 0; i < 8; ++i) lrow[i] = lrow[i] * alpha[i] + bsum[i];
#pragma unroll
    for (int t = 0; t < 4; ++t)
#pragma unroll
      for (int i = 0; i < 8; ++i) o[t][i] = o[t][i] * alpha[i];
    // --- re-fragment P (C layout -> A layout) through LDS ---
#pragma unroll
    for (int t = 0; t < 2; ++t)
#pragma unroll
      for (int i = 0; i < 8; ++i)
        pl[(rowB + i) * 32 + 16 * t + cn] = (bf16)sc[t][i];
    asm volatile("s_wait_dscnt 0x0" ::: "memory");  // intra-wave LDS RAW
    v16bf pf = load_frag_l(pl, 0, lane);
    // --- O += P V ---
#pragma unroll
    for (int t = 0; t < 4; ++t) {
      v16bf vf = load_frag_g(Vh, S_LEN, 16 * t, k0, lane);
      o[t] = wmma_bf16f32(pf, vf, o[t]);
    }
  }
  // --- normalize, write bf16 attn output [s][NH*HD] ---
#pragma unroll
  for (int i = 0; i < 8; ++i) lrow[i] = 1.f / lrow[i];
#pragma unroll
  for (int t = 0; t < 4; ++t)
#pragma unroll
    for (int i = 0; i < 8; ++i) {
      int r = q0 + rowB + i;
      int d = 16 * t + cn;
      Ob[(size_t)r * DIM_ + h * HD + d] = (bf16)(o[t][i] * lrow[i]);
    }
}

// ---------------------------------------------------------------------------
// Host orchestration
// ---------------------------------------------------------------------------
extern "C" void kernel_launch(void* const* d_in, const int* in_sizes, int n_in,
                              void* d_out, int out_size, void* d_ws,
                              size_t ws_size, hipStream_t stream) {
  (void)in_sizes; (void)n_in; (void)out_size; (void)ws_size;
  const float* x  = (const float*)d_in[0];
  const float* fc = (const float*)d_in[1];
  const float* fs = (const float*)d_in[2];
  // d_in[3] = mask: intentionally unread (causal mask computed inline)
  const float* wq = (const float*)d_in[4];
  const float* wk = (const float*)d_in[5];
  const float* wv = (const float*)d_in[6];
  const float* wo = (const float*)d_in[7];
  float* out = (float*)d_out;

  char* ws = (char*)d_ws;
  size_t off = 0;
  auto alloc = [&](size_t bytes) -> void* {
    void* p = ws + off;
    off += (bytes + 255) & ~(size_t)255;
    return p;
  };
  bf16* Xb   = (bf16*)alloc((size_t)S_LEN * DIM_ * 2);          // 8 MB
  bf16* WqT  = (bf16*)alloc((size_t)DIM_ * DIM_ * 2);           // 8 MB
  bf16* WkT  = (bf16*)alloc((size_t)(NKV * HD) * DIM_ * 2);     // 2 MB
  bf16* WvT  = (bf16*)alloc((size_t)(NKV * HD) * DIM_ * 2);     // 2 MB
  bf16* WoT  = (bf16*)alloc((size_t)DIM_ * DIM_ * 2);           // 8 MB
  float* Qf  = (float*)alloc((size_t)S_LEN * DIM_ * 4);         // 16 MB
  float* Kf  = (float*)alloc((size_t)S_LEN * (NKV * HD) * 4);   // 4 MB
  float* Vf  = (float*)alloc((size_t)S_LEN * (NKV * HD) * 4);   // 4 MB
  bf16* Qbp  = (bf16*)alloc((size_t)NH * S_LEN * HD * 2);       // 8 MB
  bf16* Kbp  = (bf16*)alloc((size_t)NKV * S_LEN * HD * 2);      // 2 MB
  bf16* Vtb  = (bf16*)alloc((size_t)NKV * HD * S_LEN * 2);      // 2 MB
  bf16* Ab   = (bf16*)alloc((size_t)S_LEN * DIM_ * 2);          // 8 MB

  const int nx = S_LEN * DIM_;
  k_cvt_bf16<<<nx / 256, 256, 0, stream>>>(x, Xb, nx);
  k_tr_cvt<<<(DIM_ * DIM_) / 256, 256, 0, stream>>>(wq, WqT, DIM_, DIM_);
  k_tr_cvt<<<(DIM_ * NKV * HD) / 256, 256, 0, stream>>>(wk, WkT, DIM_, NKV * HD);
  k_tr_cvt<<<(DIM_ * NKV * HD) / 256, 256, 0, stream>>>(wv, WvT, DIM_, NKV * HD);
  k_tr_cvt<<<(DIM_ * DIM_) / 256, 256, 0, stream>>>(wo, WoT, DIM_, DIM_);

  // QKV projections (bf16 WMMA, f32 accumulate); block tile 256x64
  k_gemm<<<dim3(S_LEN / 256, DIM_ / 64), 256, 0, stream>>>(Xb, WqT, Qf, S_LEN, DIM_, DIM_);
  k_gemm<<<dim3(S_LEN / 256, (NKV * HD) / 64), 256, 0, stream>>>(Xb, WkT, Kf, S_LEN, NKV * HD, DIM_);
  k_gemm<<<dim3(S_LEN / 256, (NKV * HD) / 64), 256, 0, stream>>>(Xb, WvT, Vf, S_LEN, NKV * HD, DIM_);

  // RoPE + per-head repack
  k_rope_pack<<<(NH * S_LEN * (HD / 2)) / 256, 256, 0, stream>>>(Qf, Qbp, fc, fs, NH);
  k_rope_pack<<<(NKV * S_LEN * (HD / 2)) / 256, 256, 0, stream>>>(Kf, Kbp, fc, fs, NKV);
  k_pack_vT<<<(NKV * HD * S_LEN) / 256, 256, 0, stream>>>(Vf, Vtb);

  // Flash attention: 32 heads * 128 q-tiles, 8 wave-tiles per block
  k_flash<<<dim3((NH * (S_LEN / 16)) / 8), 256, 0, stream>>>(Qbp, Kbp, Vtb, Ab);

  // Output projection straight into d_out (f32)
  k_gemm<<<dim3(S_LEN / 256, DIM_ / 64), 256, 0, stream>>>(Ab, WoT, out, S_LEN, DIM_, DIM_);
}
